// AAEncoder_90331752170013
// MI455X (gfx1250) — compile-verified, matrix operand order checked by reading, not run on hardware
//
#include <hip/hip_runtime.h>
#include <hip/hip_bf16.h>

#define NN 40000
#define EE 640000
#define DD 128

typedef __bf16 bf16;
typedef __attribute__((ext_vector_type(16))) __bf16 v16bf;
typedef __attribute__((ext_vector_type(8)))  float  v8f;

union Frag16 { v16bf v; uint4 q[2]; };
union FragC  { v8f  v; float f[8]; };

// ---------- helpers ----------

__device__ __forceinline__ float rowsum16(float x) {
  x += __shfl_xor(x, 1, 16);
  x += __shfl_xor(x, 2, 16);
  x += __shfl_xor(x, 4, 16);
  x += __shfl_xor(x, 8, 16);
  return x;
}

__device__ __forceinline__ float rowsum8(float x) {
  x += __shfl_xor(x, 1, 8);
  x += __shfl_xor(x, 2, 8);
  x += __shfl_xor(x, 4, 8);
  return x;
}

// One wave computes a 16x16 tile of  Y = A(16xK) * W^T  with bf16 WMMA.
__device__ __forceinline__ v8f wmma_acc(const bf16* As, int lda,
                                        const bf16* Wb, int ldw, int kcount,
                                        v8f acc) {
  const int lane = threadIdx.x & 31;
  const int nl = lane & 15, g = lane >> 4;
  const bf16* arow = As + nl * lda;
  const bf16* wrow = Wb + (size_t)nl * ldw;
  for (int kb = 0; kb < kcount; kb += 32) {
    Frag16 a, b;
    a.q[0] = *(const uint4*)(arow + kb + 8 * g);
    a.q[1] = *(const uint4*)(arow + kb + 16 + 8 * g);
    b.q[0] = *(const uint4*)(wrow + kb + 8 * g);
    b.q[1] = *(const uint4*)(wrow + kb + 16 + 8 * g);
    acc = __builtin_amdgcn_wmma_f32_16x16x32_bf16(false, a.v, false, b.v,
                                                  (short)0, acc, false, false);
  }
  return acc;
}

// Two M-tiles (rows 0-15 and 16-31) sharing one B-fragment load per K step.
__device__ __forceinline__ void wmma_pair(const bf16* As, int lda,
                                          const bf16* Wb, int ldw,
                                          v8f& acc0, v8f& acc1) {
  const int lane = threadIdx.x & 31;
  const int nl = lane & 15, g = lane >> 4;
  const bf16* arow0 = As + nl * lda;
  const bf16* arow1 = As + (16 + nl) * lda;
  const bf16* wrow = Wb + (size_t)nl * ldw;
  for (int kb = 0; kb < 128; kb += 32) {
    Frag16 a0, a1, b;
    b.q[0]  = *(const uint4*)(wrow + kb + 8 * g);
    b.q[1]  = *(const uint4*)(wrow + kb + 16 + 8 * g);
    a0.q[0] = *(const uint4*)(arow0 + kb + 8 * g);
    a0.q[1] = *(const uint4*)(arow0 + kb + 16 + 8 * g);
    a1.q[0] = *(const uint4*)(arow1 + kb + 8 * g);
    a1.q[1] = *(const uint4*)(arow1 + kb + 16 + 8 * g);
    acc0 = __builtin_amdgcn_wmma_f32_16x16x32_bf16(false, a0.v, false, b.v,
                                                   (short)0, acc0, false, false);
    acc1 = __builtin_amdgcn_wmma_f32_16x16x32_bf16(false, a1.v, false, b.v,
                                                   (short)0, acc1, false, false);
  }
}

// Store one wave's 16x16 f32 tile (+bias) into a row-major buffer.
__device__ __forceinline__ void tile_store(float* Out, int ldo, int colbase,
                                           v8f acc, const float* bias,
                                           bool accumulate) {
  const int lane = threadIdx.x & 31;
  const int nl = lane & 15, g = lane >> 4;
  FragC c; c.v = acc;
  float bv = bias ? bias[colbase + nl] : 0.f;
  for (int i = 0; i < 8; ++i) {
    float val = c.f[i] + bv;
    float* p = Out + (size_t)(i + 8 * g) * ldo + colbase + nl;
    if (accumulate) *p += val; else *p = val;
  }
}

// LayerNorm over 16 rows x 128 cols (256 threads: 16 per row).
__device__ __forceinline__ void ln_block(const float* X, int ldx,
                                         const float* gam, const float* bet,
                                         bf16* OutB, int ldb,
                                         float* OutF, int ldf, bool do_relu) {
  const int r = threadIdx.x >> 4;
  const int c0 = threadIdx.x & 15;
  float vals[8]; float s = 0.f, s2 = 0.f;
  for (int i = 0; i < 8; ++i) {
    float v = X[r * ldx + c0 + 16 * i];
    vals[i] = v; s += v; s2 += v * v;
  }
  s = rowsum16(s); s2 = rowsum16(s2);
  float mean = s * (1.f / 128.f);
  float var  = s2 * (1.f / 128.f) - mean * mean;
  float rstd = rsqrtf(var + 1e-5f);
  for (int i = 0; i < 8; ++i) {
    int c = c0 + 16 * i;
    float y = (vals[i] - mean) * rstd * gam[c] + bet[c];
    if (do_relu) y = fmaxf(y, 0.f);
    if (OutB) OutB[r * ldb + c] = (bf16)y;
    if (OutF) OutF[r * ldf + c] = y;
  }
}

// LayerNorm over 32 rows x 128 cols (256 threads: 8 per row).
__device__ __forceinline__ void ln_block32(const float* X, int ldx,
                                           const float* gam, const float* bet,
                                           bf16* OutB, int ldb, bool do_relu) {
  const int r = threadIdx.x >> 3;
  const int c0 = threadIdx.x & 7;
  float vals[16]; float s = 0.f, s2 = 0.f;
  for (int i = 0; i < 16; ++i) {
    float v = X[r * ldx + c0 + 8 * i];
    vals[i] = v; s += v; s2 += v * v;
  }
  s = rowsum8(s); s2 = rowsum8(s2);
  float mean = s * (1.f / 128.f);
  float var  = s2 * (1.f / 128.f) - mean * mean;
  float rstd = rsqrtf(var + 1e-5f);
  for (int i = 0; i < 16; ++i) {
    int c = c0 + 8 * i;
    float y = (vals[i] - mean) * rstd * gam[c] + bet[c];
    if (do_relu) y = fmaxf(y, 0.f);
    OutB[r * ldb + c] = (bf16)y;
  }
}

__device__ __forceinline__ void atomicMaxFloat(float* addr, float val) {
  int* ia = (int*)addr;
  int old = __float_as_int(*addr);
  while (__int_as_float(old) < val) {
    int assumed = old;
    old = atomicCAS(ia, assumed, __float_as_int(val));
    if (old == assumed) break;
  }
}

// ---------- kernels ----------

__global__ void k_f32_to_bf16(const float* __restrict__ src,
                              bf16* __restrict__ dst, int n) {
  int i = blockIdx.x * 256 + threadIdx.x;
  if (i < n) dst[i] = (bf16)src[i];
}

__global__ void k_init(float* __restrict__ msg, float* __restrict__ denom,
                       float* __restrict__ amax) {
  int i = blockIdx.x * 256 + threadIdx.x;
  if (i < NN * DD) msg[i] = 0.f;
  if (i < NN * 8) { denom[i] = 0.f; amax[i] = -3.0e38f; }
}

// Center embedding (SingleInputEmbedding) + bos mask + q projection.
__global__ __launch_bounds__(256)
void k_node(const float* __restrict__ x, const float* __restrict__ rot,
            const unsigned char* __restrict__ bos_mask,
            const float* __restrict__ bos_token, const int* __restrict__ t_ptr,
            const float* ce1_w, const float* ce1_b,
            const float* ln1g, const float* ln1b,
            const bf16* ce2_w, const float* ce2_b,
            const float* ln2g, const float* ln2b,
            const bf16* ce3_w, const float* ce3_b,
            const float* ln3g, const float* ln3b,
            const bf16* qw, const float* qb,
            float* __restrict__ ce_out, float* __restrict__ q_out) {
  __shared__ __align__(16) bf16 As[16][136];
  __shared__ float Sf[16][132];
  __shared__ float sxr[16][2];

  const int tid = threadIdx.x;
  const int r0 = blockIdx.x * 16;
  const int wave = tid >> 5;

  if (tid < 16) {
    int row = r0 + tid;
    float x0 = x[row * 2 + 0], x1 = x[row * 2 + 1];
    const float* R = rot + (size_t)row * 4;
    sxr[tid][0] = x0 * R[0] + x1 * R[2];
    sxr[tid][1] = x0 * R[1] + x1 * R[3];
  }
  __syncthreads();
  { // 2 -> 128 linear
    int r = tid >> 4, c0 = tid & 15;
    float a0 = sxr[r][0], a1 = sxr[r][1];
    for (int i = 0; i < 8; ++i) {
      int c = c0 + 16 * i;
      Sf[r][c] = ce1_w[c * 2] * a0 + ce1_w[c * 2 + 1] * a1 + ce1_b[c];
    }
  }
  __syncthreads();
  ln_block(&Sf[0][0], 132, ln1g, ln1b, &As[0][0], 136, nullptr, 0, true);
  __syncthreads();
  { v8f acc = {};
    acc = wmma_acc(&As[0][0], 136, ce2_w + (size_t)(wave * 16) * DD, DD, DD, acc);
    tile_store(&Sf[0][0], 132, wave * 16, acc, ce2_b, false); }
  __syncthreads();
  ln_block(&Sf[0][0], 132, ln2g, ln2b, &As[0][0], 136, nullptr, 0, true);
  __syncthreads();
  { v8f acc = {};
    acc = wmma_acc(&As[0][0], 136, ce3_w + (size_t)(wave * 16) * DD, DD, DD, acc);
    tile_store(&Sf[0][0], 132, wave * 16, acc, ce3_b, false); }
  __syncthreads();
  { // final LN + bos select, write ce + bf16 for q GEMM
    int r = tid >> 4, c0 = tid & 15;
    int row = r0 + r;
    float vals[8]; float s = 0.f, s2 = 0.f;
    for (int i = 0; i < 8; ++i) {
      float v = Sf[r][c0 + 16 * i]; vals[i] = v; s += v; s2 += v * v;
    }
    s = rowsum16(s); s2 = rowsum16(s2);
    float mean = s * (1.f / 128.f), var = s2 * (1.f / 128.f) - mean * mean;
    float rstd = rsqrtf(var + 1e-5f);
    bool m = bos_mask[row] != 0;
    int t = *t_ptr;
    for (int i = 0; i < 8; ++i) {
      int c = c0 + 16 * i;
      float y = (vals[i] - mean) * rstd * ln3g[c] + ln3b[c];
      if (m) y = bos_token[t * DD + c];
      ce_out[(size_t)row * DD + c] = y;
      As[r][c] = (bf16)y;
    }
  }
  __syncthreads();
  { v8f acc = {};
    acc = wmma_acc(&As[0][0], 136, qw + (size_t)(wave * 16) * DD, DD, DD, acc);
    tile_store(q_out + (size_t)r0 * DD, DD, wave * 16, acc, qb, false); }
}

// Fused edge pipeline, 32 edges per block: rotate -> MultipleInputEmbedding
// -> k,v -> q.k dots.  Each wave computes two M-tiles sharing B fragments.
__global__ __launch_bounds__(256)
void k_edge(const float* __restrict__ x, const float* __restrict__ rot,
            const int* __restrict__ eidx, const float* __restrict__ eattr,
            const float* nx1_w, const float* nx1_b,
            const float* nxlng, const float* nxlnb,
            const bf16* nx2_w, const float* nx2_b,
            const float* ne1_w, const float* ne1_b,
            const float* nelng, const float* nelnb,
            const bf16* ne2_w, const float* ne2_b,
            const float* ag1g, const float* ag1b,
            const bf16* ag_w, const float* ag_b,
            const float* ag2g, const float* ag2b,
            const bf16* kw, const float* kb_,
            const bf16* vw, const float* vb_,
            const float* __restrict__ qbuf,
            float* __restrict__ alpha, float* __restrict__ amax,
            bf16* __restrict__ vout) {
  __shared__ __align__(16) bf16 As[32][136];
  __shared__ __align__(16) bf16 Nbr[32][136];
  __shared__ float Sf[32][132];
  __shared__ float Hs[32][128];
  __shared__ int   sDst[32];
  __shared__ float sxr[32][2], ser[32][2];

  const int tid = threadIdx.x;
  const int e0 = blockIdx.x * 32;
  const int wave = tid >> 5;
  const int col = wave * 16;

  { // warm the weight matrices into WGP$/L2 (128B per thread per matrix)
    const bf16* wm[5] = {nx2_w, ne2_w, ag_w, kw, vw};
    for (int m = 0; m < 5; ++m) __builtin_prefetch(wm[m] + tid * 64, 0, 3);
  }
  if (tid < 32) {
    int e = e0 + tid;
    int d = eidx[e];
    int s = eidx[EE + e];
    sDst[tid] = d;
    const float* R = rot + (size_t)s * 4;
    float x0 = x[s * 2], x1 = x[s * 2 + 1];
    sxr[tid][0] = x0 * R[0] + x1 * R[2];
    sxr[tid][1] = x0 * R[1] + x1 * R[3];
    float a0 = eattr[(size_t)e * 2], a1 = eattr[(size_t)e * 2 + 1];
    ser[tid][0] = a0 * R[0] + a1 * R[2];
    ser[tid][1] = a0 * R[1] + a1 * R[3];
  }
  __syncthreads();
  { // nx path: 2 -> 128
    int r = tid >> 3, c0 = tid & 7;
    float a0 = sxr[r][0], a1 = sxr[r][1];
    for (int i = 0; i < 16; ++i) {
      int c = c0 + 8 * i;
      Sf[r][c] = nx1_w[c * 2] * a0 + nx1_w[c * 2 + 1] * a1 + nx1_b[c];
    }
  }
  __syncthreads();
  ln_block32(&Sf[0][0], 132, nxlng, nxlnb, &As[0][0], 136, true);
  __syncthreads();
  { v8f a0 = {}, a1 = {};
    wmma_pair(&As[0][0], 136, nx2_w + (size_t)col * DD, DD, a0, a1);
    tile_store(&Hs[0][0], 128, col, a0, nx2_b, false);
    tile_store(&Hs[16][0], 128, col, a1, nx2_b, false); }
  __syncthreads();
  { // ne path: 2 -> 128
    int r = tid >> 3, c0 = tid & 7;
    float a0 = ser[r][0], a1 = ser[r][1];
    for (int i = 0; i < 16; ++i) {
      int c = c0 + 8 * i;
      Sf[r][c] = ne1_w[c * 2] * a0 + ne1_w[c * 2 + 1] * a1 + ne1_b[c];
    }
  }
  __syncthreads();
  ln_block32(&Sf[0][0], 132, nelng, nelnb, &As[0][0], 136, true);
  __syncthreads();
  { v8f a0 = {}, a1 = {};
    wmma_pair(&As[0][0], 136, ne2_w + (size_t)col * DD, DD, a0, a1);
    tile_store(&Hs[0][0], 128, col, a0, ne2_b, true);     // hx + he
    tile_store(&Hs[16][0], 128, col, a1, ne2_b, true); }
  __syncthreads();
  ln_block32(&Hs[0][0], 128, ag1g, ag1b, &As[0][0], 136, true);
  __syncthreads();
  { v8f a0 = {}, a1 = {};
    wmma_pair(&As[0][0], 136, ag_w + (size_t)col * DD, DD, a0, a1);
    tile_store(&Sf[0][0], 132, col, a0, ag_b, false);
    tile_store(&Sf[16][0], 132, col, a1, ag_b, false); }
  __syncthreads();
  ln_block32(&Sf[0][0], 132, ag2g, ag2b, &Nbr[0][0], 136, false);
  __syncthreads();
  { v8f a0 = {}, a1 = {};  // k = nbr @ k_w^T + k_b
    wmma_pair(&Nbr[0][0], 136, kw + (size_t)col * DD, DD, a0, a1);
    tile_store(&Sf[0][0], 132, col, a0, kb_, false);
    tile_store(&Sf[16][0], 132, col, a1, kb_, false); }
  __syncthreads();
  { // alpha[e][h] = q.k / 4, atomic segment-max (all 256 threads)
    int er = tid >> 3, h = tid & 7;
    int dst = sDst[er];
    const float* qrow = qbuf + (size_t)dst * DD + h * 16;
    float dot = 0.f;
    for (int d = 0; d < 16; ++d) dot += qrow[d] * Sf[er][h * 16 + d];
    dot *= 0.25f;
    alpha[(size_t)(e0 + er) * 8 + h] = dot;
    atomicMaxFloat(&amax[(size_t)dst * 8 + h], dot);
  }
  __syncthreads();
  { v8f a0 = {}, a1 = {};  // v = nbr @ v_w^T + v_b
    wmma_pair(&Nbr[0][0], 136, vw + (size_t)col * DD, DD, a0, a1);
    tile_store(&Sf[0][0], 132, col, a0, vb_, false);
    tile_store(&Sf[16][0], 132, col, a1, vb_, false); }
  __syncthreads();
  { int r = tid >> 3, c0 = tid & 7;
    for (int i = 0; i < 16; ++i) {
      int c = c0 + 8 * i;
      vout[(size_t)(e0 + r) * DD + c] = (bf16)Sf[r][c];
    } }
}

// exp(alpha - amax[dst]) scatter: 8 bf16 v-elements per thread (one b128
// load, always within one head), f32 atomics for denom and ex*v.
__global__ __launch_bounds__(256)
void k_scatter(const int* __restrict__ eidx, const float* __restrict__ alpha,
               const float* __restrict__ amax, const bf16* __restrict__ vbuf,
               float* __restrict__ denom, float* __restrict__ msg) {
  size_t tg = (size_t)blockIdx.x * 256 + threadIdx.x;
  int e  = (int)(tg >> 4);
  int d0 = ((int)tg & 15) * 8;
  int h  = d0 >> 4;
  int dst = eidx[e];
  float a  = alpha[(size_t)e * 8 + h];
  float am = amax[(size_t)dst * 8 + h];
  if (am < -1e30f) am = 0.f;
  float ex = __expf(a - am);
  union { uint4 q; bf16 e8[8]; } vv;
  vv.q = *(const uint4*)(vbuf + (size_t)e * DD + d0);
  float* mrow = msg + (size_t)dst * DD + d0;
  for (int i = 0; i < 8; ++i) atomicAdd(&mrow[i], ex * (float)vv.e8[i]);
  if ((d0 & 15) == 0) atomicAdd(&denom[(size_t)dst * 8 + h], ex);
}

// Node update: gated blend, out proj, norm1, MLP (chunked K), norm2.
__global__ __launch_bounds__(256)
void k_update(const float* __restrict__ ce_in,
              const float* __restrict__ msgacc, const float* __restrict__ denom,
              const bf16* ihw, const float* ihb,
              const bf16* hhw, const float* hhb,
              const bf16* sw, const float* sb,
              const bf16* ow, const float* ob,
              const float* n1g, const float* n1b,
              const bf16* m1w, const float* m1b,
              const bf16* m2w, const float* m2b,
              const float* n2g, const float* n2b,
              float* __restrict__ out) {
  __shared__ __align__(16) bf16 As[16][136];
  __shared__ __align__(16) bf16 As2[16][136];
  __shared__ float Sf[16][132];
  __shared__ float Sf2[16][132];
  __shared__ float Sce[16][128];
  __shared__ float Smsg[16][128];

  const int tid = threadIdx.x;
  const int r0 = blockIdx.x * 16;
  const int wave = tid >> 5;

  { int r = tid >> 4, c0 = tid & 15; int row = r0 + r;
    for (int i = 0; i < 8; ++i) {
      int c = c0 + 16 * i;
      float ce = ce_in[(size_t)row * DD + c];
      float dn = denom[(size_t)row * 8 + (c >> 4)];
      float m = msgacc[(size_t)row * DD + c] / (dn + 1e-16f);
      Sce[r][c] = ce; Smsg[r][c] = m;
      As[r][c] = (bf16)m;     // msg for ih
      As2[r][c] = (bf16)ce;   // ce for hh/self
    } }
  __syncthreads();
  // ih(msg) -> Sf ; hh(ce) += Sf ; self(ce) -> Sf2 (each wave owns its cols)
  { v8f acc = {};
    acc = wmma_acc(&As[0][0], 136, ihw + (size_t)(wave * 16) * DD, DD, DD, acc);
    tile_store(&Sf[0][0], 132, wave * 16, acc, ihb, false); }
  { v8f acc = {};
    acc = wmma_acc(&As2[0][0], 136, hhw + (size_t)(wave * 16) * DD, DD, DD, acc);
    tile_store(&Sf[0][0], 132, wave * 16, acc, hhb, true); }
  { v8f acc = {};
    acc = wmma_acc(&As2[0][0], 136, sw + (size_t)(wave * 16) * DD, DD, DD, acc);
    tile_store(&Sf2[0][0], 132, wave * 16, acc, sb, false); }
  __syncthreads();
  { // gate & blended output -> As (input of out-proj)
    int r = tid >> 4, c0 = tid & 15;
    for (int i = 0; i < 8; ++i) {
      int c = c0 + 16 * i;
      float gate = 1.f / (1.f + __expf(-Sf[r][c]));
      float m = Smsg[r][c];
      As[r][c] = (bf16)(m + gate * (Sf2[r][c] - m));
    } }
  __syncthreads();
  { v8f acc = {};
    acc = wmma_acc(&As[0][0], 136, ow + (size_t)(wave * 16) * DD, DD, DD, acc);
    tile_store(&Sf[0][0], 132, wave * 16, acc, ob, false); }
  __syncthreads();
  { int r = tid >> 4, c0 = tid & 15;
    for (int i = 0; i < 8; ++i) { int c = c0 + 16 * i; Sf[r][c] += Sce[r][c]; } }
  __syncthreads();
  ln_block(&Sf[0][0], 132, n1g, n1b, &As[0][0], 136, &Sce[0][0], 128, false);
  __syncthreads();
  // MLP 128 -> 512 -> 128, K chunked by 128; ff stays in registers.
  v8f ff = {};
  for (int ch = 0; ch < 4; ++ch) {
    { v8f acc = {};
      acc = wmma_acc(&As[0][0], 136,
                     m1w + (size_t)(ch * 128 + wave * 16) * DD, DD, DD, acc);
      tile_store(&Sf[0][0], 132, wave * 16, acc, m1b + ch * 128, false); }
    __syncthreads();
    { int r = tid >> 4, c0 = tid & 15;
      for (int i = 0; i < 8; ++i) {
        int c = c0 + 16 * i;
        As2[r][c] = (bf16)fmaxf(Sf[r][c], 0.f);
      } }
    __syncthreads();
    ff = wmma_acc(&As2[0][0], 136,
                  m2w + (size_t)(wave * 16) * 512 + ch * 128, 512, DD, ff);
    __syncthreads();
  }
  tile_store(&Sf[0][0], 132, wave * 16, ff, m2b, false);
  __syncthreads();
  { int r = tid >> 4, c0 = tid & 15;
    for (int i = 0; i < 8; ++i) { int c = c0 + 16 * i; Sf[r][c] += Sce[r][c]; } }
  __syncthreads();
  { // final LN -> out
    int r = tid >> 4, c0 = tid & 15; int row = r0 + r;
    float vals[8]; float s = 0.f, s2 = 0.f;
    for (int i = 0; i < 8; ++i) {
      float v = Sf[r][c0 + 16 * i]; vals[i] = v; s += v; s2 += v * v;
    }
    s = rowsum16(s); s2 = rowsum16(s2);
    float mean = s * (1.f / 128.f), var = s2 * (1.f / 128.f) - mean * mean;
    float rstd = rsqrtf(var + 1e-5f);
    for (int i = 0; i < 8; ++i) {
      int c = c0 + 16 * i;
      out[(size_t)row * DD + c] = (vals[i] - mean) * rstd * n2g[c] + n2b[c];
    } }
}

// ---------- launcher ----------

extern "C" void kernel_launch(void* const* d_in, const int* in_sizes, int n_in,
                              void* d_out, int out_size, void* d_ws, size_t ws_size,
                              hipStream_t stream) {
  (void)in_sizes; (void)out_size; (void)ws_size;
  const float* x     = (const float*)d_in[0];
  const int*   eidx  = (const int*)d_in[1];
  const float* eattr = (const float*)d_in[2];
  const unsigned char* bos = (const unsigned char*)d_in[3];
  const int*   t_ptr = (const int*)d_in[4];
  const float* rot   = (const float*)d_in[5];

  const float* P[56];
  for (int i = 6; i < n_in && i < 6 + 56; ++i) P[i - 6] = (const float*)d_in[i];
  enum {
    ce1_w = 0, ce1_b, ce_ln1_g, ce_ln1_b, ce2_w, ce2_b, ce_ln2_g, ce_ln2_b,
    ce3_w, ce3_b, ce_ln3_g, ce_ln3_b,
    nx1_w, nx1_b, nx_ln_g, nx_ln_b, nx2_w, nx2_b,
    ne1_w, ne1_b, ne_ln_g, ne_ln_b, ne2_w, ne2_b,
    ag_ln1_g, ag_ln1_b, ag_w, ag_b, ag_ln2_g, ag_ln2_b,
    q_w, q_b, k_w, k_b, v_w, v_b, self_w, self_b,
    ih_w, ih_b, hh_w, hh_b, out_w, out_b,
    norm1_g, norm1_b, norm2_g, norm2_b,
    mlp1_w, mlp1_b, mlp2_w, mlp2_b, bos_token
  };

  char* ws = (char*)d_ws;
  size_t off = 0;
  auto alloc = [&](size_t bytes) {
    void* p = ws + off;
    off += bytes; off = (off + 255) & ~(size_t)255;
    return p;
  };

  bf16* bCE2 = (bf16*)alloc(DD * DD * 2);
  bf16* bCE3 = (bf16*)alloc(DD * DD * 2);
  bf16* bNX2 = (bf16*)alloc(DD * DD * 2);
  bf16* bNE2 = (bf16*)alloc(DD * DD * 2);
  bf16* bAG  = (bf16*)alloc(DD * DD * 2);
  bf16* bQ   = (bf16*)alloc(DD * DD * 2);
  bf16* bK   = (bf16*)alloc(DD * DD * 2);
  bf16* bV   = (bf16*)alloc(DD * DD * 2);
  bf16* bSF  = (bf16*)alloc(DD * DD * 2);
  bf16* bIH  = (bf16*)alloc(DD * DD * 2);
  bf16* bHH  = (bf16*)alloc(DD * DD * 2);
  bf16* bOW  = (bf16*)alloc(DD * DD * 2);
  bf16* bM1  = (bf16*)alloc(512 * DD * 2);
  bf16* bM2  = (bf16*)alloc(DD * 512 * 2);
  float* wCE   = (float*)alloc((size_t)NN * DD * 4);
  float* wQ    = (float*)alloc((size_t)NN * DD * 4);
  float* wAl   = (float*)alloc((size_t)EE * 8 * 4);
  float* wAmax = (float*)alloc((size_t)NN * 8 * 4);
  float* wDen  = (float*)alloc((size_t)NN * 8 * 4);
  float* wMsg  = (float*)alloc((size_t)NN * DD * 4);
  bf16*  wV    = (bf16*)alloc((size_t)EE * DD * 2);

  auto conv = [&](const float* s, bf16* d, int n) {
    k_f32_to_bf16<<<(n + 255) / 256, 256, 0, stream>>>(s, d, n);
  };
  conv(P[ce2_w], bCE2, DD * DD);   conv(P[ce3_w], bCE3, DD * DD);
  conv(P[nx2_w], bNX2, DD * DD);   conv(P[ne2_w], bNE2, DD * DD);
  conv(P[ag_w],  bAG,  DD * DD);   conv(P[q_w],   bQ,   DD * DD);
  conv(P[k_w],   bK,   DD * DD);   conv(P[v_w],   bV,   DD * DD);
  conv(P[self_w], bSF, DD * DD);   conv(P[ih_w],  bIH,  DD * DD);
  conv(P[hh_w],  bHH,  DD * DD);   conv(P[out_w], bOW,  DD * DD);
  conv(P[mlp1_w], bM1, 512 * DD);  conv(P[mlp2_w], bM2, DD * 512);

  k_init<<<(NN * DD + 255) / 256, 256, 0, stream>>>(wMsg, wDen, wAmax);

  k_node<<<NN / 16, 256, 0, stream>>>(
      x, rot, bos, P[bos_token], t_ptr,
      P[ce1_w], P[ce1_b], P[ce_ln1_g], P[ce_ln1_b],
      bCE2, P[ce2_b], P[ce_ln2_g], P[ce_ln2_b],
      bCE3, P[ce3_b], P[ce_ln3_g], P[ce_ln3_b],
      bQ, P[q_b], wCE, wQ);

  k_edge<<<EE / 32, 256, 0, stream>>>(
      x, rot, eidx, eattr,
      P[nx1_w], P[nx1_b], P[nx_ln_g], P[nx_ln_b], bNX2, P[nx2_b],
      P[ne1_w], P[ne1_b], P[ne_ln_g], P[ne_ln_b], bNE2, P[ne2_b],
      P[ag_ln1_g], P[ag_ln1_b], bAG, P[ag_b], P[ag_ln2_g], P[ag_ln2_b],
      bK, P[k_b], bV, P[v_b], wQ, wAl, wAmax, wV);

  k_scatter<<<(int)(((size_t)EE * 16) / 256), 256, 0, stream>>>(
      eidx, wAl, wAmax, wV, wDen, wMsg);

  k_update<<<NN / 16, 256, 0, stream>>>(
      wCE, wMsg, wDen,
      bIH, P[ih_b], bHH, P[hh_b], bSF, P[self_b], bOW, P[out_b],
      P[norm1_g], P[norm1_b], bM1, P[mlp1_b], bM2, P[mlp2_b],
      P[norm2_g], P[norm2_b], (float*)d_out);
}
